// AdaptiveSpareCrossAttention_29240137351479
// MI455X (gfx1250) — matrix operand order repaired and checked by reading, not compiled, not run on hardware
//
#include <hip/hip_runtime.h>

// ---------------------------------------------------------------------------
// CDNA5 (gfx1250) wave32 WMMA implementation of the blended cross-attention.
// GEMMs: TDM (tensor_load_to_lds) double-buffered LDS staging + v_wmma bf16.
// Attention: flash-style online softmax + relu^2 path, all WMMA.
// ---------------------------------------------------------------------------

typedef __attribute__((ext_vector_type(16))) __bf16 v16bf;
typedef __attribute__((ext_vector_type(8)))  __bf16 v8bf;
typedef __attribute__((ext_vector_type(8)))  float  v8f;
typedef __attribute__((ext_vector_type(4)))  unsigned int v4u;
typedef __attribute__((ext_vector_type(8)))  int v8i;
typedef __attribute__((ext_vector_type(4)))  int v4i;

#define B_    2
#define SEQ_  2048
#define DIM_  1024
#define H_    16
#define DH_   64
#define ROWS_ (B_ * SEQ_)   // 4096

#if __has_builtin(__builtin_amdgcn_tensor_load_to_lds)
#define USE_TDM 1
#else
#define USE_TDM 0
#endif

// ---- WMMA fragment loaders (ISA 7.12.2 layouts, row-major source) ----------

// A operand 16x32 bf16: lane = row; K chunks at kk+8*half and kk+16+8*half.
__device__ __forceinline__ v16bf frag_a_load(const __bf16* row, int kk, int lane) {
  const int h = lane >> 4;
  v8bf lo = *reinterpret_cast<const v8bf*>(row + kk + h * 8);
  v8bf hi = *reinterpret_cast<const v8bf*>(row + kk + 16 + h * 8);
  v16bf a;
#pragma unroll
  for (int i = 0; i < 8; ++i) { a[i] = lo[i]; a[i + 8] = hi[i]; }
  return a;
}

// B operand 32x16 bf16 (Bt rows): lane = column's Bt row; K chunks at
// kk+8*half and kk+16+8*half.
__device__ __forceinline__ v16bf frag_b_load(const __bf16* row, int kk, int lane) {
  const int h = lane >> 4;
  v8bf lo = *reinterpret_cast<const v8bf*>(row + kk + h * 8);
  v8bf hi = *reinterpret_cast<const v8bf*>(row + kk + 16 + h * 8);
  v16bf b;
#pragma unroll
  for (int i = 0; i < 8; ++i) { b[i] = lo[i]; b[i + 8] = hi[i]; }
  return b;
}

__device__ __forceinline__ float rmax16(float x) {
#pragma unroll
  for (int m = 1; m < 16; m <<= 1) x = fmaxf(x, __shfl_xor(x, m, 32));
  return x;
}
__device__ __forceinline__ float rsum16(float x) {
#pragma unroll
  for (int m = 1; m < 16; m <<= 1) x += __shfl_xor(x, m, 32);
  return x;
}

// ---- Tensor Data Mover: 2D bf16 tile (tile_d0 x tile_d1 rows) -> LDS -------
// D# packing per CDNA5 ISA 8.3/8.4. global ptr = tile start, row stride in
// elements, compact row-major destination in LDS.

__device__ __forceinline__ unsigned lds_offset_of(const void* p) {
  return (unsigned)(unsigned long long)p;  // low 32 bits of generic = LDS offset
}

#if USE_TDM
__device__ __forceinline__ void tdm_load_2d_bf16(unsigned lds_off, const __bf16* gptr,
                                                 unsigned tensor_d0, unsigned tensor_d1,
                                                 unsigned long long d0_stride,
                                                 unsigned tile_d0, unsigned tile_d1) {
  unsigned long long ga = (unsigned long long)gptr;
  v4u g0;
  g0[0] = 1u;                                          // count=1, no gather
  g0[1] = lds_off;                                     // lds_addr (bytes)
  g0[2] = (unsigned)(ga & 0xFFFFFFFFu);                // global_addr[31:0]
  g0[3] = (unsigned)((ga >> 32) & 0x01FFFFFFu) | 0x80000000u; // addr[56:32]|type=2
  v8i g1;
  g1[0] = (int)(1u << 16);                             // data_size=1 (2 bytes)
  g1[1] = (int)((tensor_d0 & 0xFFFFu) << 16);          // tensor_dim0[15:0]
  g1[2] = (int)((tensor_d0 >> 16) | ((tensor_d1 & 0xFFFFu) << 16));
  g1[3] = (int)((tensor_d1 >> 16) | ((tile_d0 & 0xFFFFu) << 16));
  g1[4] = (int)(tile_d1 & 0xFFFFu);                    // tile_dim1, tile_dim2=0
  g1[5] = (int)(d0_stride & 0xFFFFFFFFull);            // dim0_stride[31:0]
  g1[6] = (int)((d0_stride >> 32) & 0xFFFFull);        // dim0_stride[47:32]
  g1[7] = 0;
  v4i z4 = {0, 0, 0, 0};
  v8i z8 = {0, 0, 0, 0, 0, 0, 0, 0};
  __builtin_amdgcn_tensor_load_to_lds(g0, g1, z4, z4, z8, 0);
}
#endif

// ---- Prep kernels ----------------------------------------------------------

__global__ void to_bf16_kernel(const float* __restrict__ src, __bf16* __restrict__ dst, int n) {
  int i = blockIdx.x * blockDim.x + threadIdx.x;
  if (i < n) dst[i] = (__bf16)src[i];
}

// src: f32 [K x N] (in x out) -> dst: bf16 [N x K]
__global__ void transpose_bf16_kernel(const float* __restrict__ src, __bf16* __restrict__ dst,
                                      int K, int N) {
  size_t i = (size_t)blockIdx.x * blockDim.x + threadIdx.x;
  if (i >= (size_t)K * N) return;
  int k = (int)(i % K);
  int j = (int)(i / K);
  dst[i] = (__bf16)src[(size_t)k * N + j];
}

// KV bf16 [B*N, 2C] -> Vt bf16 [B,H,Dh,N]
__global__ void vt_transpose_kernel(const __bf16* __restrict__ KV, __bf16* __restrict__ Vt) {
  size_t i = (size_t)blockIdx.x * blockDim.x + threadIdx.x;
  if (i >= (size_t)B_ * H_ * DH_ * SEQ_) return;
  int n = (int)(i % SEQ_);
  size_t r = i / SEQ_;
  int d = (int)(r % DH_); r /= DH_;
  int h = (int)(r % H_);
  int b = (int)(r / H_);
  Vt[i] = KV[(size_t)(b * SEQ_ + n) * (2 * DIM_) + DIM_ + h * DH_ + d];
}

// ---- Cooperative NT GEMM: block = 8 waves = 128x64 tile --------------------
// TDM double-buffers A(128x32) and B(64x32) bf16 slabs into LDS; waves consume
// fragments via ds_load_b128 and run v_wmma_f32_16x16x32_bf16.

template <bool F32OUT>
__device__ __forceinline__ void gemm_nt_body(const __bf16* __restrict__ A,
                                             const __bf16* __restrict__ Bt,
                                             const float* __restrict__ bias,
                                             __bf16* __restrict__ Cb,
                                             float* __restrict__ Cf,
                                             int M, int N, int K) {
  __shared__ __align__(16) __bf16 ldsA[2][128 * 32];   // 2 x 8 KiB
  __shared__ __align__(16) __bf16 ldsB[2][64 * 32];    // 2 x 4 KiB

  const int wid  = threadIdx.x >> 5;
  const int lane = threadIdx.x & 31;
  const int ln = lane & 15, hh = lane >> 4;
  const int nb0 = blockIdx.x * 64;
  const int mb0 = blockIdx.y * 128;
  const bool issuer = (wid == 0);

#if USE_TDM
  if (issuer) {
    tdm_load_2d_bf16(lds_offset_of(ldsA[0]), A + (size_t)mb0 * K, K, M, K, 32, 128);
    tdm_load_2d_bf16(lds_offset_of(ldsB[0]), Bt + (size_t)nb0 * K, K, N, K, 32, 64);
  }
#endif

  v8f acc[4] = {};
  for (int kk = 0; kk < K; kk += 32) {
    const int cur = (kk >> 5) & 1;
#if USE_TDM
    if (issuer) {
      if (kk + 32 < K) {
        tdm_load_2d_bf16(lds_offset_of(ldsA[cur ^ 1]), A + (size_t)mb0 * K + kk + 32,
                         K, M, K, 32, 128);
        tdm_load_2d_bf16(lds_offset_of(ldsB[cur ^ 1]), Bt + (size_t)nb0 * K + kk + 32,
                         K, N, K, 32, 64);
        __builtin_amdgcn_s_wait_tensorcnt((short)2);   // cur slab done, next in flight
      } else {
        __builtin_amdgcn_s_wait_tensorcnt((short)0);
      }
    }
#else
    // Fallback: cooperative 16B copies (256 threads: A 512 chunks, B 256 chunks)
    {
      int t = threadIdx.x;
#pragma unroll
      for (int c = 0; c < 2; ++c) {
        int idx = t + c * 256;            // 0..511 -> row idx/4, chunk idx%4
        int rowA = idx >> 2, ch = idx & 3;
        *(v8bf*)&ldsA[cur][rowA * 32 + ch * 8] =
            *(const v8bf*)(A + ((size_t)(mb0 + rowA)) * K + kk + ch * 8);
      }
      int rowB = t >> 2, chB = t & 3;
      *(v8bf*)&ldsB[cur][rowB * 32 + chB * 8] =
          *(const v8bf*)(Bt + ((size_t)(nb0 + rowB)) * K + kk + chB * 8);
    }
#endif
    __syncthreads();

    const __bf16* arow = &ldsA[cur][(wid * 16 + ln) * 32];
    v16bf af = frag_a_load(arow, 0, lane);
#pragma unroll
    for (int j = 0; j < 4; ++j) {
      const __bf16* brow = &ldsB[cur][(j * 16 + ln) * 32];
      v16bf bf = frag_b_load(brow, 0, lane);
      acc[j] = __builtin_amdgcn_wmma_f32_16x16x32_bf16(false, af, false, bf,
                                                       (short)0, acc[j], false, false);
    }
    __syncthreads();   // protect buffer reuse two iterations ahead
  }

  const int mrow = mb0 + wid * 16;
#pragma unroll
  for (int j = 0; j < 4; ++j)
#pragma unroll
    for (int v = 0; v < 8; ++v) {
      int col = nb0 + j * 16 + ln;
      size_t o = (size_t)(mrow + v + 8 * hh) * N + col;
      if (F32OUT) Cf[o] = acc[j][v] + bias[col];
      else        Cb[o] = (__bf16)acc[j][v];
    }
}

__global__ __launch_bounds__(256)
void gemm_nt_bf16_kernel(const __bf16* __restrict__ A, const __bf16* __restrict__ Bt,
                         __bf16* __restrict__ C, int M, int N, int K) {
  gemm_nt_body<false>(A, Bt, nullptr, C, nullptr, M, N, K);
}

__global__ __launch_bounds__(256)
void gemm_nt_f32_bias_kernel(const __bf16* __restrict__ A, const __bf16* __restrict__ Bt,
                             const float* __restrict__ bias, float* __restrict__ C,
                             int M, int N, int K) {
  gemm_nt_body<true>(A, Bt, bias, nullptr, C, M, N, K);
}

// ---- Attention: one wave per (b, h, 16-row block), flash-style -------------

__global__ __launch_bounds__(32)
void attn_kernel(const __bf16* __restrict__ Q, const __bf16* __restrict__ KV,
                 const __bf16* __restrict__ Vt, const float* __restrict__ w,
                 __bf16* __restrict__ O) {
  __shared__ __align__(16) __bf16 lds[2][16 * 32];   // P tiles C-layout -> A-layout
  const int lane = threadIdx.x & 31;
  const int ln = lane & 15, hh = lane >> 4;
  const int mtiles = SEQ_ / 16;
  const int mb = (blockIdx.x % mtiles) * 16;
  const int bh = blockIdx.x / mtiles;
  const int h = bh % H_;
  const int b = bh / H_;
  const float scale = 0.125f;

  const float w0 = w[0], w1 = w[1];
  const float wm = fmaxf(w0, w1);
  const float e0 = __expf(w0 - wm), e1 = __expf(w1 - wm);
  const float wn0 = e0 / (e0 + e1), wn1 = e1 / (e0 + e1);

  const __bf16* qrow = Q + ((size_t)(b * SEQ_ + mb + ln) * DIM_ + h * DH_);
  const v16bf qa0 = frag_a_load(qrow, 0, lane);
  const v16bf qa1 = frag_a_load(qrow, 32, lane);

  v8f acc0[4] = {};   // exp path (online rescaled)
  v8f acc1[4] = {};   // relu^2 path
  float mrun[8], lsum[8];
#pragma unroll
  for (int v = 0; v < 8; ++v) { mrun[v] = -__builtin_inff(); lsum[v] = 0.f; }

  for (int nb = 0; nb < SEQ_; nb += 32) {
    // prefetch next n-tile's K rows and Vt stripe (global_prefetch_b8)
    if (nb + 32 < SEQ_) {
      __builtin_prefetch(KV + ((size_t)(b * SEQ_ + nb + 32 + lane) * (2 * DIM_) + h * DH_), 0, 3);
      __builtin_prefetch(Vt + ((size_t)((b * H_ + h) * DH_ + (lane & 63)) * SEQ_ + nb + 32), 0, 3);
    }

    v8f s[2];
#pragma unroll
    for (int t = 0; t < 2; ++t) {
      const __bf16* krow = KV + ((size_t)(b * SEQ_ + nb + t * 16 + ln) * (2 * DIM_) + h * DH_);
      v16bf kb0 = frag_b_load(krow, 0, lane);
      v16bf kb1 = frag_b_load(krow, 32, lane);
      v8f z = {};
      z = __builtin_amdgcn_wmma_f32_16x16x32_bf16(false, qa0, false, kb0, (short)0, z, false, false);
      z = __builtin_amdgcn_wmma_f32_16x16x32_bf16(false, qa1, false, kb1, (short)0, z, false, false);
#pragma unroll
      for (int v = 0; v < 8; ++v) s[t][v] = z[v] * scale;
    }

    float mnew[8], r[8];
#pragma unroll
    for (int v = 0; v < 8; ++v) {
      float tm = rmax16(fmaxf(s[0][v], s[1][v]));
      mnew[v] = fmaxf(mrun[v], tm);
      r[v] = __expf(mrun[v] - mnew[v]);
      mrun[v] = mnew[v];
    }

#pragma unroll
    for (int t = 0; t < 2; ++t)
#pragma unroll
      for (int v = 0; v < 8; ++v) {
        float sv = s[t][v];
        float p0 = __expf(sv - mnew[v]);
        float p1 = sv > 0.f ? sv * sv : 0.f;
        lds[0][(v + 8 * hh) * 32 + t * 16 + ln] = (__bf16)p0;
        lds[1][(v + 8 * hh) * 32 + t * 16 + ln] = (__bf16)p1;
        s[t][v] = p0;
      }
#pragma unroll
    for (int v = 0; v < 8; ++v) {
      float ts = rsum16(s[0][v] + s[1][v]);
      lsum[v] = lsum[v] * r[v] + ts;
    }
    __syncthreads();

    const v16bf pf0 = frag_a_load(&lds[0][ln * 32], 0, lane);
    const v16bf pf1 = frag_a_load(&lds[1][ln * 32], 0, lane);

#pragma unroll
    for (int j = 0; j < 4; ++j)
#pragma unroll
      for (int v = 0; v < 8; ++v) acc0[j][v] *= r[v];

#pragma unroll
    for (int j = 0; j < 4; ++j) {
      const __bf16* vrow = Vt + ((size_t)((b * H_ + h) * DH_ + j * 16 + ln) * SEQ_ + nb);
      v16bf vb = frag_b_load(vrow, 0, lane);
      acc0[j] = __builtin_amdgcn_wmma_f32_16x16x32_bf16(false, pf0, false, vb, (short)0, acc0[j], false, false);
      acc1[j] = __builtin_amdgcn_wmma_f32_16x16x32_bf16(false, pf1, false, vb, (short)0, acc1[j], false, false);
    }
    __syncthreads();
  }

#pragma unroll
  for (int j = 0; j < 4; ++j)
#pragma unroll
    for (int v = 0; v < 8; ++v) {
      float o = wn0 * acc0[j][v] / lsum[v] + wn1 * acc1[j][v];
      O[(size_t)(b * SEQ_ + mb + v + 8 * hh) * DIM_ + h * DH_ + j * 16 + ln] = (__bf16)o;
    }
}

// ---------------------------------------------------------------------------

extern "C" void kernel_launch(void* const* d_in, const int* in_sizes, int n_in,
                              void* d_out, int out_size, void* d_ws, size_t ws_size,
                              hipStream_t stream) {
  (void)in_sizes; (void)n_in; (void)out_size;
  const float* x1    = (const float*)d_in[0];
  const float* x2    = (const float*)d_in[1];
  const float* Wq    = (const float*)d_in[2];
  const float* Wkv   = (const float*)d_in[3];
  const float* Wproj = (const float*)d_in[4];
  const float* bproj = (const float*)d_in[5];
  const float* w     = (const float*)d_in[6];
  float* out = (float*)d_out;

  char* ws = (char*)d_ws;
  size_t off = 0;
  auto alloc_bf16 = [&](size_t elems) {
    __bf16* p = (__bf16*)(ws + off);
    off += elems * sizeof(__bf16);
    return p;
  };
  __bf16* X1   = alloc_bf16((size_t)ROWS_ * DIM_);
  __bf16* X2   = alloc_bf16((size_t)ROWS_ * DIM_);
  __bf16* Wqt  = alloc_bf16((size_t)DIM_ * DIM_);
  __bf16* Wkvt = alloc_bf16((size_t)2 * DIM_ * DIM_);
  __bf16* Wpt  = alloc_bf16((size_t)DIM_ * DIM_);
  __bf16* Qb   = alloc_bf16((size_t)ROWS_ * DIM_);
  __bf16* KVb  = alloc_bf16((size_t)ROWS_ * 2 * DIM_);
  __bf16* Vt   = alloc_bf16((size_t)B_ * H_ * DH_ * SEQ_);
  __bf16* AO   = alloc_bf16((size_t)ROWS_ * DIM_);
  if (off > ws_size) return;   // needs ~67 MiB

  const int nIn = ROWS_ * DIM_;
  to_bf16_kernel<<<(nIn + 255) / 256, 256, 0, stream>>>(x1, X1, nIn);
  to_bf16_kernel<<<(nIn + 255) / 256, 256, 0, stream>>>(x2, X2, nIn);
  transpose_bf16_kernel<<<(DIM_ * DIM_ + 255) / 256, 256, 0, stream>>>(Wq, Wqt, DIM_, DIM_);
  transpose_bf16_kernel<<<(2 * DIM_ * DIM_ + 255) / 256, 256, 0, stream>>>(Wkv, Wkvt, DIM_, 2 * DIM_);
  transpose_bf16_kernel<<<(DIM_ * DIM_ + 255) / 256, 256, 0, stream>>>(Wproj, Wpt, DIM_, DIM_);

  // Q = X1 @ Wq : blocks of 128x64 output tiles
  gemm_nt_bf16_kernel<<<dim3(DIM_ / 64, ROWS_ / 128), 256, 0, stream>>>(X1, Wqt, Qb, ROWS_, DIM_, DIM_);
  // KV = X2 @ Wkv
  gemm_nt_bf16_kernel<<<dim3(2 * DIM_ / 64, ROWS_ / 128), 256, 0, stream>>>(X2, Wkvt, KVb, ROWS_, 2 * DIM_, DIM_);

  vt_transpose_kernel<<<(B_ * H_ * DH_ * SEQ_) / 256, 256, 0, stream>>>(KVb, Vt);

  attn_kernel<<<B_ * H_ * (SEQ_ / 16), 32, 0, stream>>>(Qb, KVb, Vt, w, AO);

  gemm_nt_f32_bias_kernel<<<dim3(DIM_ / 64, ROWS_ / 128), 256, 0, stream>>>(AO, Wpt, bproj, out, ROWS_, DIM_, DIM_);
}
